// SchNetInteractionBlock_78743930404961
// MI455X (gfx1250) — matrix operand order repaired and checked by reading, not compiled, not run on hardware
//
#include <hip/hip_runtime.h>
#include <stdint.h>

// ---------------------------------------------------------------------------
// SchNet interaction block, fused for MI455X (gfx1250, wave32, WMMA bf16)
// ---------------------------------------------------------------------------

typedef __attribute__((ext_vector_type(16))) __bf16   v16bf;
typedef __attribute__((ext_vector_type(2)))  __bf16   v2bf;
typedef __attribute__((ext_vector_type(8)))  float    v8f;
typedef __attribute__((ext_vector_type(4)))  uint32_t u32x4;

#define NF 128            // N_BASIS == N_FILTERS == 128
#define N_ATOMS_C 50000
#define N_PAIRS_C 1600000
#define N_RBF_C 20

// ---- native bf16 conversion (packed where the builtin exists) -------------
static __device__ __forceinline__ uint32_t pack2bf(float x, float y) {
#if __has_builtin(__builtin_amdgcn_cvt_pk_bf16_f32)
  union { v2bf v; uint32_t u; } r;
  r.v = __builtin_amdgcn_cvt_pk_bf16_f32(x, y);
  return r.u;
#else
  union { uint32_t u; __bf16 h[2]; } r;
  r.h[0] = (__bf16)x;
  r.h[1] = (__bf16)y;
  return r.u;
#endif
}

static __device__ __forceinline__ uint16_t f2bf1(float x) {
  union { __bf16 h; uint16_t s; } r;
  r.h = (__bf16)x;
  return r.s;
}

// shifted softplus: softplus(x) - log(2), numerically-stable form
static __device__ __forceinline__ float sspf(float x) {
  return fmaxf(x, 0.0f) + __logf(1.0f + __expf(-fabsf(x))) - 0.69314718055994530942f;
}

static __device__ __forceinline__ v8f wmma_bf16(v16bf a, v16bf b, v8f c) {
  // (neg_a, A, neg_b, B, c_mod, C, reuse_a, reuse_b)
  return __builtin_amdgcn_wmma_f32_16x16x32_bf16(false, a, false, b, (short)0, c,
                                                 false, false);
}

// --- B fragment: pre-swizzled bf16 weights, 8 dwords per lane per fragment ---
// frag = kstep*8 + nblk ; element (lane,i,p): N = nblk*16+(lane&15),
// K = kstep*32 + (i>=4)*16 + (lane>>4)*8 + (i&3)*2 + p   (ISA 7.12.2 wave32 B layout)
static __device__ __forceinline__ v16bf load_B(const uint32_t* w, int frag, int lane) {
  union { v16bf v; u32x4 q[2]; } r;
  const u32x4* p = (const u32x4*)(w + (size_t)(frag * 32 + lane) * 8);
  r.q[0] = p[0];
  r.q[1] = p[1];
  return r.v;
}

// --- A fragment (16x32 bf16) from row-major f32 [*,128] matrix ---
// element (lane,i,p): M = lane&15, K = k0 + (i>=4)*16 + (lane>>4)*8 + (i&3)*2 + p
static __device__ __forceinline__ v16bf load_A_g128(const float* base, int row0,
                                                    int k0, int lane) {
  union { v16bf v; uint32_t u[8]; } r;
  const float* p = base + (size_t)(row0 + (lane & 15)) * NF + k0 + ((lane >> 4) << 3);
#pragma unroll
  for (int i = 0; i < 8; ++i) {
    int k = ((i >= 4) ? 16 : 0) + ((i & 3) << 1);
    float2 f = *(const float2*)(p + k);        // 8B aligned: k even, row 512B aligned
    r.u[i] = pack2bf(f.x, f.y);
  }
  return r.v;
}

// --- A fragment from f_ij [*,20] f32, zero-padded to K=32 ---
static __device__ __forceinline__ v16bf load_A_fij(const float* f_ij, int e0, int lane) {
  union { v16bf v; uint32_t u[8]; } r;
  const float* p = f_ij + (size_t)(e0 + (lane & 15)) * N_RBF_C;
  int kb = (lane >> 4) << 3;
#pragma unroll
  for (int i = 0; i < 8; ++i) {
    int k = kb + ((i >= 4) ? 16 : 0) + ((i & 3) << 1);
    float fx = (k < N_RBF_C)     ? p[k]     : 0.0f;
    float fy = (k + 1 < N_RBF_C) ? p[k + 1] : 0.0f;
    r.u[i] = pack2bf(fx, fy);
  }
  return r.v;
}

// --- A fragment from a 16x128 bf16 LDS tile (C->A transpose restage) ---
static __device__ __forceinline__ v16bf load_A_lds(const uint16_t* t, int k0, int lane) {
  union { v16bf v; uint32_t u[8]; } r;
  const uint16_t* p = t + (lane & 15) * NF + k0 + ((lane >> 4) << 3);
#pragma unroll
  for (int i = 0; i < 8; ++i) {
    int k = ((i >= 4) ? 16 : 0) + ((i & 3) << 1);
    r.u[i] = *(const uint32_t*)(p + k);        // 2 packed bf16, 4B aligned
  }
  return r.v;
}

// ---------------------------------------------------------------------------
// Kernel 0a: zero the accumulator buffer (agg lives in d_out)
// ---------------------------------------------------------------------------
__global__ void zero_f32(float* p, int n) {
  int stride = gridDim.x * blockDim.x;
  for (int i = blockIdx.x * blockDim.x + threadIdx.x; i < n; i += stride)
    p[i] = 0.0f;
}

// ---------------------------------------------------------------------------
// Kernel 0b: swizzle f32 weights [fan_in x 128] into bf16 B fragments
// ---------------------------------------------------------------------------
__global__ void swizzle_w(const float* __restrict__ W, uint32_t* __restrict__ dst,
                          int fan_in, int nksteps) {
  int tid = blockIdx.x * blockDim.x + threadIdx.x;
  int total = nksteps * 8 * 32 * 8;
  if (tid >= total) return;
  int i    = tid & 7;
  int lane = (tid >> 3) & 31;
  int frag = tid >> 8;
  int kstep = frag >> 3, nblk = frag & 7;
  int n = nblk * 16 + (lane & 15);
  int k = kstep * 32 + ((i >= 4) ? 16 : 0) + ((lane >> 4) << 3) + ((i & 3) << 1);
  float lo = (k < fan_in)     ? W[(size_t)k * NF + n]       : 0.0f;
  float hi = (k + 1 < fan_in) ? W[(size_t)(k + 1) * NF + n] : 0.0f;
  dst[(size_t)(frag * 32 + lane) * 8 + i] = pack2bf(lo, hi);
}

// ---------------------------------------------------------------------------
// Kernel 1: h = x @ W_in + b_in   (one wave computes a 16x128 tile)
// ---------------------------------------------------------------------------
__global__ void node_linear(const float* __restrict__ X, const uint32_t* __restrict__ Ws,
                            const float* __restrict__ bias, float* __restrict__ D,
                            int ntiles) {
  int lane = threadIdx.x & 31;
  int wave = (blockIdx.x * blockDim.x + threadIdx.x) >> 5;
  if (wave >= ntiles) return;
  int row0 = wave << 4;
  v8f acc[8];
#pragma unroll
  for (int n = 0; n < 8; ++n) acc[n] = (v8f){0.f,0.f,0.f,0.f,0.f,0.f,0.f,0.f};
#pragma unroll
  for (int ks = 0; ks < 4; ++ks) {
    v16bf a = load_A_g128(X, row0, ks * 32, lane);
#pragma unroll
    for (int n = 0; n < 8; ++n)
      acc[n] = wmma_bf16(a, load_B(Ws, ks * 8 + n, lane), acc[n]);
  }
  int mo = (lane >> 4) << 3;
  int cl = lane & 15;
#pragma unroll
  for (int n = 0; n < 8; ++n) {
    float b = bias[n * 16 + cl];
#pragma unroll
    for (int v = 0; v < 8; ++v)
      D[(size_t)(row0 + mo + v) * NF + n * 16 + cl] = acc[n][v] + b;
  }
}

// ---------------------------------------------------------------------------
// Kernel 2: fused edge pipeline.
//   Wij = (ssp(f_ij@Wf1 + bf1) @ Wf2 + bf2) * rcut
//   agg[idx_i] += h[idx_j] * Wij         (f32 atomics; gather/scatter in L2)
// One wave = 16 edges; 4 waves/WG; LDS restages the ssp(t) tile C->A layout.
// h-row gathers are prefetched at wave start so their latency hides under the
// 40 WMMAs (global_prefetch_b8).
// ---------------------------------------------------------------------------
__global__ void edge_fused(const float* __restrict__ f_ij, const float* __restrict__ rcut,
                           const int* __restrict__ idx_i, const int* __restrict__ idx_j,
                           const uint32_t* __restrict__ Wf1s, const float* __restrict__ bf1,
                           const uint32_t* __restrict__ Wf2s, const float* __restrict__ bf2,
                           const float* __restrict__ h, float* __restrict__ agg,
                           int ntiles) {
  __shared__ __align__(16) uint16_t tbuf[4][16 * NF];
  __shared__ float srcut[4][16];
  __shared__ int   sii[4][16];
  __shared__ int   sij[4][16];

  int lane = threadIdx.x & 31;
  int wv   = threadIdx.x >> 5;
  int wave = blockIdx.x * 4 + wv;
  bool valid = wave < ntiles;
  int tile = valid ? wave : 0;
  int e0 = tile << 4;
  int mo = (lane >> 4) << 3;
  int cl = lane & 15;

  if (lane < 16) {
    srcut[wv][lane] = rcut[e0 + lane];
    sii[wv][lane]   = idx_i[e0 + lane];
    sij[wv][lane]   = idx_j[e0 + lane];
  }

  // Prefetch the 16 gathered h rows (512B each): lane covers row (lane&15),
  // halves of the wave prefetch the two 256B halves of each row.
  {
    int jrow = idx_j[e0 + cl];
    const float* hp = h + (size_t)jrow * NF + ((lane >> 4) << 6);
    __builtin_prefetch(hp, 0, 3);
    __builtin_prefetch(hp + 32, 0, 3);
  }

  // GEMM 1 (K = 32, padded from 20): t = ssp(f_ij @ Wf1 + bf1) -> LDS (bf16)
  v16bf a0 = load_A_fij(f_ij, e0, lane);
#pragma unroll
  for (int n = 0; n < 8; ++n) {
    v8f c = (v8f){0.f,0.f,0.f,0.f,0.f,0.f,0.f,0.f};
    c = wmma_bf16(a0, load_B(Wf1s, n, lane), c);
    float b = bf1[n * 16 + cl];
#pragma unroll
    for (int v = 0; v < 8; ++v)
      tbuf[wv][(mo + v) * NF + n * 16 + cl] = f2bf1(sspf(c[v] + b));
  }
  __syncthreads();

  // GEMM 2 (K = 128): Wij = t @ Wf2 + bf2
  v8f acc[8];
#pragma unroll
  for (int n = 0; n < 8; ++n) acc[n] = (v8f){0.f,0.f,0.f,0.f,0.f,0.f,0.f,0.f};
#pragma unroll
  for (int ks = 0; ks < 4; ++ks) {
    v16bf a = load_A_lds(&tbuf[wv][0], ks * 32, lane);
#pragma unroll
    for (int n = 0; n < 8; ++n)
      acc[n] = wmma_bf16(a, load_B(Wf2s, ks * 8 + n, lane), acc[n]);
  }

  if (!valid) return;

  // cutoff scale, gather h[idx_j], scatter-add into agg[idx_i]
#pragma unroll
  for (int n = 0; n < 8; ++n) {
    float b = bf2[n * 16 + cl];
    int col = n * 16 + cl;
#pragma unroll
    for (int v = 0; v < 8; ++v) {
      int m = mo + v;
      float wij = (acc[n][v] + b) * srcut[wv][m];
      float hv  = h[(size_t)sij[wv][m] * NF + col];
      atomicAdd(&agg[(size_t)sii[wv][m] * NF + col], hv * wij);
    }
  }
}

// ---------------------------------------------------------------------------
// Kernel 3: out = ssp(agg @ Wo1 + bo1) @ Wo2 + bo2   (in-place over agg/d_out;
// each 16-row tile depends only on its own agg rows, read before any store)
// ---------------------------------------------------------------------------
__global__ void output_fused(const float* __restrict__ agg,
                             const uint32_t* __restrict__ Wo1s, const float* __restrict__ bo1,
                             const uint32_t* __restrict__ Wo2s, const float* __restrict__ bo2,
                             float* __restrict__ out, int ntiles) {
  __shared__ __align__(16) uint16_t tbuf[4][16 * NF];
  int lane = threadIdx.x & 31;
  int wv   = threadIdx.x >> 5;
  int wave = blockIdx.x * 4 + wv;
  bool valid = wave < ntiles;
  int tile = valid ? wave : 0;
  int row0 = tile << 4;
  int mo = (lane >> 4) << 3;
  int cl = lane & 15;

  v8f acc[8];
#pragma unroll
  for (int n = 0; n < 8; ++n) acc[n] = (v8f){0.f,0.f,0.f,0.f,0.f,0.f,0.f,0.f};
#pragma unroll
  for (int ks = 0; ks < 4; ++ks) {
    v16bf a = load_A_g128(agg, row0, ks * 32, lane);
#pragma unroll
    for (int n = 0; n < 8; ++n)
      acc[n] = wmma_bf16(a, load_B(Wo1s, ks * 8 + n, lane), acc[n]);
  }
#pragma unroll
  for (int n = 0; n < 8; ++n) {
    float b = bo1[n * 16 + cl];
#pragma unroll
    for (int v = 0; v < 8; ++v)
      tbuf[wv][(mo + v) * NF + n * 16 + cl] = f2bf1(sspf(acc[n][v] + b));
  }
  __syncthreads();

#pragma unroll
  for (int n = 0; n < 8; ++n) acc[n] = (v8f){0.f,0.f,0.f,0.f,0.f,0.f,0.f,0.f};
#pragma unroll
  for (int ks = 0; ks < 4; ++ks) {
    v16bf a = load_A_lds(&tbuf[wv][0], ks * 32, lane);
#pragma unroll
    for (int n = 0; n < 8; ++n)
      acc[n] = wmma_bf16(a, load_B(Wo2s, ks * 8 + n, lane), acc[n]);
  }

  if (!valid) return;
#pragma unroll
  for (int n = 0; n < 8; ++n) {
    float b = bo2[n * 16 + cl];
#pragma unroll
    for (int v = 0; v < 8; ++v)
      out[(size_t)(row0 + mo + v) * NF + n * 16 + cl] = acc[n][v] + b;
  }
}

// ---------------------------------------------------------------------------
extern "C" void kernel_launch(void* const* d_in, const int* in_sizes, int n_in,
                              void* d_out, int out_size, void* d_ws, size_t ws_size,
                              hipStream_t stream) {
  (void)in_sizes; (void)n_in; (void)ws_size;

  const float* x     = (const float*)d_in[0];
  const float* f_ij  = (const float*)d_in[1];
  const float* rcut  = (const float*)d_in[2];
  const float* W_in  = (const float*)d_in[3];
  const float* b_in  = (const float*)d_in[4];
  const float* Wf1   = (const float*)d_in[5];
  const float* bf1   = (const float*)d_in[6];
  const float* Wf2   = (const float*)d_in[7];
  const float* bf2   = (const float*)d_in[8];
  const float* Wo1   = (const float*)d_in[9];
  const float* bo1   = (const float*)d_in[10];
  const float* Wo2   = (const float*)d_in[11];
  const float* bo2   = (const float*)d_in[12];
  const int*   idx_i = (const int*)d_in[13];
  const int*   idx_j = (const int*)d_in[14];
  float* out = (float*)d_out;   // doubles as agg accumulator (zeroed below)

  // workspace layout: h [50000x128 f32], then pre-swizzled bf16 weight frags
  float*    h    = (float*)d_ws;
  uint32_t* wb   = (uint32_t*)((char*)d_ws + (size_t)N_ATOMS_C * NF * sizeof(float));
  uint32_t* Wins = wb;              // 4 ksteps * 8 nblk * 32 lanes * 8 = 8192 dwords
  uint32_t* Wf1s = Wins + 8192;     // 1 kstep  -> 2048 dwords
  uint32_t* Wf2s = Wf1s + 2048;
  uint32_t* Wo1s = Wf2s + 8192;
  uint32_t* Wo2s = Wo1s + 8192;

  zero_f32<<<2048, 256, 0, stream>>>(out, out_size);
  swizzle_w<<<32, 256, 0, stream>>>(W_in, Wins, NF, 4);
  swizzle_w<<<8,  256, 0, stream>>>(Wf1,  Wf1s, N_RBF_C, 1);
  swizzle_w<<<32, 256, 0, stream>>>(Wf2,  Wf2s, NF, 4);
  swizzle_w<<<32, 256, 0, stream>>>(Wo1,  Wo1s, NF, 4);
  swizzle_w<<<32, 256, 0, stream>>>(Wo2,  Wo2s, NF, 4);

  int node_tiles = N_ATOMS_C / 16;   // 3125
  node_linear<<<(node_tiles + 3) / 4, 128, 0, stream>>>(x, Wins, b_in, h, node_tiles);

  int edge_tiles = N_PAIRS_C / 16;   // 100000 (divisible by 4 waves/WG)
  edge_fused<<<edge_tiles / 4, 128, 0, stream>>>(f_ij, rcut, idx_i, idx_j,
                                                 Wf1s, bf1, Wf2s, bf2, h, out, edge_tiles);

  output_fused<<<(node_tiles + 3) / 4, 128, 0, stream>>>(out, Wo1s, bo1, Wo2s, bo2,
                                                         out, node_tiles);
}
